// GnnModel_25254407700577
// MI455X (gfx1250) — matrix operand order repaired and checked by reading, not compiled, not run on hardware
//
#include <hip/hip_runtime.h>
#include <hip/hip_bf16.h>

typedef _Float16 h16 __attribute__((ext_vector_type(16)));
typedef _Float16 h8  __attribute__((ext_vector_type(8)));
typedef float    f8  __attribute__((ext_vector_type(8)));
typedef float    f4  __attribute__((ext_vector_type(4)));

__device__ __forceinline__ unsigned fkey(float f) {
    unsigned u = __float_as_uint(f);
    return (u & 0x80000000u) ? ~u : (u | 0x80000000u);
}
__device__ __forceinline__ float funkey(unsigned k) {
    unsigned u = (k & 0x80000000u) ? (k & 0x7fffffffu) : ~k;
    return __uint_as_float(u);
}
__device__ __forceinline__ void atomAddF(float* p, float v) {
    unsafeAtomicAdd(p, v);   // global_atomic_add_f32
}

// ---------------------------------------------------------------------------
// Pack a [256(out) x K(in)] f32 weight (trans=1: B[k,n] = w[n*K+k]) or a
// [K x 256] row-major f32 matrix (trans=0: B[k,n] = w[k*256+n]) into the
// WMMA f16 B-fragment order used here (contiguous k, as A is loaded):
//   fragment (kt,nt): 32 lanes x 16 halves; element e:
//   K = kt*32 + (lane>=16 ? 16 : 0) + e, N = nt*16 + (lane&15).
// ---------------------------------------------------------------------------
__global__ void pack_b(const float* __restrict__ w, _Float16* __restrict__ out,
                       int K, int trans) {
    int id = blockIdx.x * blockDim.x + threadIdx.x;
    if (id >= K * 256) return;
    int e    = id & 15;
    int lane = (id >> 4) & 31;
    int nt   = (id >> 9) & 15;
    int kt   = id >> 13;
    int k = kt * 32 + ((lane & 16) ? 16 : 0) + e;
    int n = nt * 16 + (lane & 15);
    float v = trans ? w[(size_t)n * K + k] : w[(size_t)k * 256 + n];
    out[id] = (_Float16)v;
}

// C[i,j] = sum_h Wq[h,i] * We[h,j]; cb[j] = sum_h bq[h] * We[h,j]
__global__ void cmat_kernel(const float* __restrict__ wq, const float* __restrict__ we,
                            const float* __restrict__ bq,
                            float* __restrict__ C, float* __restrict__ cb) {
    int id = blockIdx.x * blockDim.x + threadIdx.x;  // 65536 threads
    int i = id >> 8, j = id & 255;
    float s = 0.f;
    for (int h = 0; h < 256; ++h) s += wq[h * 256 + i] * we[h * 256 + j];
    C[id] = s;
    if (i == 0) {
        float sb = 0.f;
        for (int h = 0; h < 256; ++h) sb += bq[h] * we[h * 256 + j];
        cb[j] = sb;
    }
}

// ---------------------------------------------------------------------------
// WMMA GEMM: Y[N x 256] = X[N x K] * Bpacked + bias (+ addsrc), f32 accum.
// Block = 256 threads = 8 waves; two waves share one 16-row M-tile, each
// owning a 128-column half (8 accumulator tiles = 64 VGPRs -> no spills).
// Packed B is staged into LDS with GLOBAL_LOAD_ASYNC_TO_LDS_B128 (ASYNCcnt),
// in stages of <=4 K-tiles (64 KB LDS); WMMA B-fragments come from ds loads.
// ---------------------------------------------------------------------------
__global__ void gemm_wmma(const float* __restrict__ X,
                          const _Float16* __restrict__ Bp,
                          const float* __restrict__ bias,
                          const float* __restrict__ addsrc,
                          float* __restrict__ outF,
                          _Float16* __restrict__ outH,
                          int Mtiles, int K) {
    extern __shared__ _Float16 smem[];   // min(Ktiles,4)*8192 halves
    int tid  = threadIdx.x;
    int wave = tid >> 5;
    int lane = tid & 31;
    int mt    = blockIdx.x * 4 + (wave >> 1);
    int nbase = (wave & 1) * 8;           // which 128-col half
    bool active = (mt < Mtiles);          // inactive waves still do async+barriers
    int row  = mt * 16 + (lane & 15);
    int koff = (lane < 16) ? 0 : 16;
    int hi   = (lane < 16) ? 0 : 8;

    f8 acc[8] = {};
    int Ktiles  = K >> 5;
    int nStages = (Ktiles + 3) >> 2;
    for (int st = 0; st < nStages; ++st) {
        int kt0 = st * 4;
        int ktn = Ktiles - kt0; if (ktn > 4) ktn = 4;
        if (st > 0) __syncthreads();      // prior stage fully consumed
        // stage ktn*16KB of packed B into LDS (async, per-lane 16B chunks)
        int chunks = ktn * 4;             // per-thread 16B chunks
        const _Float16* gbase = Bp + (size_t)kt0 * 8192;
        for (int c = 0; c < chunks; ++c) {
            unsigned idx = (unsigned)(c * 256 + tid) * 8;  // halves
            unsigned ldsoff = (unsigned)(uintptr_t)(&smem[idx]);
            const _Float16* gp = gbase + idx;
            asm volatile("global_load_async_to_lds_b128 %0, %1, off"
                         :: "v"(ldsoff), "v"(gp) : "memory");
        }
        asm volatile("s_wait_asynccnt 0x0" ::: "memory");
        __syncthreads();

        if (active) {
            for (int ktL = 0; ktL < ktn; ++ktL) {
                int kt = kt0 + ktL;
                const float* xp = X + (size_t)row * K + kt * 32 + koff;
                f4 f0 = *(const f4*)(xp);
                f4 f1 = *(const f4*)(xp + 4);
                f4 f2 = *(const f4*)(xp + 8);
                f4 f3 = *(const f4*)(xp + 12);
                h16 a;
#pragma unroll
                for (int i = 0; i < 4; ++i) {
                    a[i]      = (_Float16)f0[i];
                    a[4 + i]  = (_Float16)f1[i];
                    a[8 + i]  = (_Float16)f2[i];
                    a[12 + i] = (_Float16)f3[i];
                }
                const _Float16* bp =
                    smem + ((size_t)(ktL * 16 + nbase) * 32 + lane) * 16;
#pragma unroll
                for (int nt = 0; nt < 8; ++nt) {
                    h8 b0 = *(const h8*)(bp + (size_t)nt * 512);
                    h8 b1 = *(const h8*)(bp + (size_t)nt * 512 + 8);
                    h16 b = __builtin_shufflevector(b0, b1, 0, 1, 2, 3, 4, 5, 6, 7,
                                                    8, 9, 10, 11, 12, 13, 14, 15);
                    acc[nt] = __builtin_amdgcn_wmma_f32_16x16x32_f16(
                        false, a, false, b, (short)0, acc[nt], false, false);
                }
            }
        }
    }
    if (!active) return;
#pragma unroll
    for (int nt = 0; nt < 8; ++nt) {
        int col = (nbase + nt) * 16 + (lane & 15);
        float bv = bias ? bias[col] : 0.f;
#pragma unroll
        for (int r = 0; r < 8; ++r) {
            int mrow = mt * 16 + r + hi;
            size_t idx = (size_t)mrow * 256 + col;
            float val = acc[nt][r] + bv;
            if (addsrc) val += addsrc[idx];
            if (outF) outF[idx] = val;
            if (outH) outH[idx] = (_Float16)val;
        }
    }
}

__global__ void copy_f4(const float* __restrict__ src, float* __restrict__ dst, long n4) {
    long i = (long)blockIdx.x * blockDim.x + threadIdx.x;
    if (i < n4) ((f4*)dst)[i] = ((const f4*)src)[i];
}

__global__ void init_red(unsigned* red) {
    if (threadIdx.x == 0) { red[0] = 0u; ((float*)red)[1] = 0.f; }
}

// wave per edge: logit = leaky_relu( (qW[src] . k[dst]) / 16 ), track global max
__global__ void edge_logits(const _Float16* __restrict__ qW, const _Float16* __restrict__ kH,
                            const int* __restrict__ src, const int* __restrict__ dst,
                            int E, float* __restrict__ logits, unsigned* __restrict__ red) {
    int gid = blockIdx.x * blockDim.x + threadIdx.x;
    int e = gid >> 5, lane = gid & 31;
    if (e >= E) return;
    int s = src[e], d = dst[e];
    h8 q = *(const h8*)(qW + (size_t)s * 256 + lane * 8);
    h8 k = *(const h8*)(kH + (size_t)d * 256 + lane * 8);
    float acc = 0.f;
#pragma unroll
    for (int j = 0; j < 8; ++j) acc += (float)q[j] * (float)k[j];
#pragma unroll
    for (int off = 16; off; off >>= 1) acc += __shfl_xor(acc, off, 32);
    if (lane == 0) {
        float l = acc * 0.0625f;
        l = (l >= 0.f) ? l : 0.01f * l;
        logits[e] = l;
        atomicMax(red, fkey(l));
    }
}

// thread per edge: w = exp(logit - max); accumulate global sum
__global__ void edge_expsum(float* __restrict__ logits, int E,
                            const unsigned* __restrict__ red, float* __restrict__ sum) {
    int e = blockIdx.x * blockDim.x + threadIdx.x;
    float m = funkey(red[0]);
    float w = 0.f;
    if (e < E) { w = expf(logits[e] - m); logits[e] = w; }
#pragma unroll
    for (int off = 16; off; off >>= 1) w += __shfl_xor(w, off, 32);
    if ((threadIdx.x & 31) == 0) atomAddF(sum, w);
}

// wave per edge: out[dst] += (w/sum) * v[src]
__global__ void edge_message(const _Float16* __restrict__ vH,
                             const int* __restrict__ src, const int* __restrict__ dst,
                             int E, const float* __restrict__ wts,
                             const float* __restrict__ sum, float* __restrict__ outAcc) {
    int gid = blockIdx.x * blockDim.x + threadIdx.x;
    int e = gid >> 5, lane = gid & 31;
    if (e >= E) return;
    float alpha = wts[e] / sum[0];
    int s = src[e], d = dst[e];
    h8 v = *(const h8*)(vH + (size_t)s * 256 + lane * 8);
    float* op = outAcc + (size_t)d * 256 + lane * 8;
#pragma unroll
    for (int j = 0; j < 8; ++j) atomAddF(op + j, alpha * (float)v[j]);
}

// ---------------------------------------------------------------------------
extern "C" void kernel_launch(void* const* d_in, const int* in_sizes, int n_in,
                              void* d_out, int out_size, void* d_ws, size_t ws_size,
                              hipStream_t stream) {
    (void)n_in; (void)out_size; (void)ws_size;
    const int FEAT[3] = {64, 128, 64};
    const float* xin[3]; int N[3];
    for (int t = 0; t < 3; ++t) {
        xin[t] = (const float*)d_in[t];
        N[t] = in_sizes[t] / FEAT[t];
    }
    const int* esrc[4]; const int* edst[4]; int EC[4];
    for (int e = 0; e < 4; ++e) {
        esrc[e] = (const int*)d_in[3 + 2 * e];
        edst[e] = (const int*)d_in[4 + 2 * e];
        EC[e] = in_sizes[3 + 2 * e];
    }
    const int ES[4] = {1, 0, 0, 2}, ED[4] = {0, 1, 2, 0};

    size_t offE[3];
    offE[0] = 0;
    offE[1] = (size_t)N[0] * 256;
    offE[2] = offE[1] + (size_t)N[1] * 256;
    size_t NTe = offE[2] + (size_t)N[2] * 256;
    int maxN = N[0]; if (N[1] > maxN) maxN = N[1]; if (N[2] > maxN) maxN = N[2];
    int maxE = EC[0]; for (int e = 1; e < 4; ++e) if (EC[e] > maxE) maxE = EC[e];

    // workspace carve
    char* p = (char*)d_ws;
    float*    xA    = (float*)p;    p += NTe * 4;
    float*    xB    = (float*)p;    p += NTe * 4;
    _Float16* kH    = (_Float16*)p; p += NTe * 2;
    _Float16* vH    = (_Float16*)p; p += NTe * 2;
    _Float16* qWH   = (_Float16*)p; p += (size_t)maxN * 256 * 2;
    float*    Cm    = (float*)p;    p += 65536 * 4;
    float*    cb    = (float*)p;    p += 256 * 4;
    _Float16* packB = (_Float16*)p; p += 65536 * 2;
    float*    lgts  = (float*)p;    p += (size_t)maxE * 4;
    unsigned* red   = (unsigned*)p;

    auto launch_pack = [&](const float* w, int K, int trans) {
        int total = K * 256;
        pack_b<<<(total + 255) / 256, 256, 0, stream>>>(w, packB, K, trans);
    };
    auto launch_gemm = [&](const float* X, const float* bias, const float* addsrc,
                           float* outF, _Float16* outH, int n, int K) {
        int Mtiles = n / 16;
        int Ktiles = K / 32;
        int stageKt = Ktiles < 4 ? Ktiles : 4;
        size_t shmem = (size_t)stageKt * 16384;   // bytes (<= 64 KB)
        gemm_wmma<<<(Mtiles + 3) / 4, 256, shmem, stream>>>(
            X, packB, bias, addsrc, outF, outH, Mtiles, K);
    };

    // stage 0: x = x_in @ lin_w^T + lin_b + emb
    for (int t = 0; t < 3; ++t) {
        const float* lw = (const float*)d_in[11 + t * 3];
        const float* lb = (const float*)d_in[12 + t * 3];
        const float* em = (const float*)d_in[13 + t * 3];
        launch_pack(lw, FEAT[t], 1);
        launch_gemm(xin[t], lb, em, xA + offE[t], nullptr, N[t], FEAT[t]);
    }

    for (int layer = 0; layer < 2; ++layer) {
        int base = 20 + layer * 22;
        const float* cur = layer ? xB : xA;
        float* nxt = layer ? (float*)d_out : xB;

        // k and v projections (f16 outputs)
        for (int t = 0; t < 3; ++t) {
            const float* kw = (const float*)d_in[base + t * 6 + 2];
            const float* kb = (const float*)d_in[base + t * 6 + 3];
            const float* vw = (const float*)d_in[base + t * 6 + 4];
            const float* vb = (const float*)d_in[base + t * 6 + 5];
            launch_pack(kw, 256, 1);
            launch_gemm(cur + offE[t], kb, nullptr, nullptr, kH + offE[t], N[t], 256);
            launch_pack(vw, 256, 1);
            launch_gemm(cur + offE[t], vb, nullptr, nullptr, vH + offE[t], N[t], 256);
        }

        // out[d] starts as clone of layer input
        long n4 = (long)(NTe / 4);
        copy_f4<<<(unsigned)((n4 + 255) / 256), 256, 0, stream>>>(cur, nxt, n4);

        // edge passes
        for (int e = 0; e < 4; ++e) {
            int s = ES[e], d = ED[e];
            const float* qw = (const float*)d_in[base + s * 6 + 0];
            const float* qb = (const float*)d_in[base + s * 6 + 1];
            const float* We = (const float*)d_in[base + 18 + e];
            cmat_kernel<<<256, 256, 0, stream>>>(qw, We, qb, Cm, cb);
            launch_pack(Cm, 256, 0);
            launch_gemm(cur + offE[s], cb, nullptr, nullptr, qWH, N[s], 256);
            init_red<<<1, 32, 0, stream>>>(red);
            int E = EC[e];
            int wt = E * 32;
            edge_logits<<<(wt + 255) / 256, 256, 0, stream>>>(
                qWH, kH + offE[d], esrc[e], edst[e], E, lgts, red);
            edge_expsum<<<(E + 255) / 256, 256, 0, stream>>>(
                lgts, E, red, (float*)(red + 1));
            edge_message<<<(wt + 255) / 256, 256, 0, stream>>>(
                vH + offE[s], esrc[e], edst[e], E, lgts, (float*)(red + 1),
                nxt + offE[d]);
        }
    }
}
